// RNNSignatureModel_62886911148777
// MI455X (gfx1250) — matrix operand order repaired and checked by reading, not compiled, not run on hardware
//
#include <hip/hip_runtime.h>
#include <hip/hip_bf16.h>
#include <cstdint>
#include <cstddef>

// ---------------------------------------------------------------------------
// RNN-Signature model for MI455X (gfx1250, wave32, WMMA).
// Pipeline:
//   1) path_kernel      : build augmented path (64,512,17)          [VALU]
//   2) cvt_pad_kernel   : f32 -> bf16 weights (K padded to 32)      [VALU]
//   3) sig_pad_kernel   : zero bf16 sig pad columns                 [VMEM]
//   4) sig_kernel       : depth-3 signatures, carry state in LDS    [LDS]
//   5) gemm_gi_kernel   : (2016x10464)x(10464x1536) bf16 WMMA GEMM,
//                         register double-buffered (software pipeline)
//   6) gru_kernel       : persistent 32-wave block, h in LDS,
//                         W_hh WMMA + fused gates + final linear    [WMMA]
// ---------------------------------------------------------------------------

#define B_SZ    32
#define S_SZ    512
#define CIN     8
#define EXTRA_C 8
#define NGROUP  2
#define C_AUG   17
#define SIGC    5219        // 17 + 289 + 4913
#define NWIN    63
#define WSTEP   8
#define WLEN    16
#define RNN     512
#define OUT_SZ  32
#define NBG     (B_SZ*NGROUP)     // 64
#define MSIG    (NWIN*B_SZ)       // 2016
#define KSIG    (NGROUP*SIGC)     // 10438
#define KPAD    10464             // 327*32  (327 odd -> pipelined loop + 1 tail)
#define NGATE   (3*RNN)           // 1536

typedef __attribute__((ext_vector_type(16))) __bf16 v16bf;
typedef __attribute__((ext_vector_type(8)))  float  v8f;

union V16 { v16bf bf; uint4 u4[2]; unsigned short us[16]; };

__device__ __forceinline__ unsigned short f2bf(float f) {
  // round-to-nearest-even f32 -> bf16
  unsigned int u = __float_as_uint(f);
  unsigned int r = u + 0x7FFFu + ((u >> 16) & 1u);
  return (unsigned short)(r >> 16);
}

// ---------------------------------------------------------------------------
// 1) augmented path: path[bg][s][0]=time, [1..8]=x, [9..16]=x@aug_W+aug_b
// ---------------------------------------------------------------------------
__global__ __launch_bounds__(256) void path_kernel(
    const float* __restrict__ x, const int* __restrict__ lengths,
    const float* __restrict__ aug_W, const float* __restrict__ aug_b,
    float* __restrict__ path) {
  int tid = blockIdx.x * blockDim.x + threadIdx.x;
  if (tid >= NBG * S_SZ) return;
  int bg = tid / S_SZ, s = tid % S_SZ;
  int b = bg / NGROUP, g = bg % NGROUP;
  int len = lengths[b];
  int si = s < (len - 1) ? s : (len - 1);
  float xv[CIN];
#pragma unroll
  for (int c = 0; c < CIN; ++c) xv[c] = x[((size_t)b * S_SZ + si) * CIN + c];
  float* p = path + (size_t)tid * C_AUG;
  p[0] = (float)s * (1.0f / (float)(S_SZ - 1));
#pragma unroll
  for (int c = 0; c < CIN; ++c) p[1 + c] = xv[c];
#pragma unroll
  for (int e = 0; e < EXTRA_C; ++e) {
    float acc = aug_b[g * EXTRA_C + e];
#pragma unroll
    for (int j = 0; j < CIN; ++j)
      acc += xv[j] * aug_W[(g * CIN + j) * EXTRA_C + e];
    p[1 + CIN + e] = acc;
  }
}

// ---------------------------------------------------------------------------
// 2) generic f32 -> bf16 row-major convert with K padding (pad -> 0)
// ---------------------------------------------------------------------------
__global__ __launch_bounds__(256) void cvt_pad_kernel(
    const float* __restrict__ src, unsigned short* __restrict__ dst,
    int rows, int kin, int kout) {
  long total = (long)rows * kout;
  for (long i = (long)blockIdx.x * blockDim.x + threadIdx.x; i < total;
       i += (long)gridDim.x * blockDim.x) {
    int r = (int)(i / kout), k = (int)(i % kout);
    dst[i] = (k < kin) ? f2bf(src[(long)r * kin + k]) : (unsigned short)0;
  }
}

// 3) zero the bf16 pad columns of the signature matrix
__global__ __launch_bounds__(256) void sig_pad_kernel(unsigned short* __restrict__ sigsA) {
  int i = blockIdx.x * blockDim.x + threadIdx.x;
  int tot = MSIG * (KPAD - KSIG);
  if (i >= tot) return;
  int r = i / (KPAD - KSIG), k = i % (KPAD - KSIG);
  sigsA[(size_t)r * KPAD + KSIG + k] = 0;
}

// ---------------------------------------------------------------------------
// 4) depth-3 truncated signature via Chen's relation; one block per window.
//    Carry state (17+289+4913 f32 = ~21 KB) lives in LDS.
// ---------------------------------------------------------------------------
__global__ __launch_bounds__(256) void sig_kernel(
    const float* __restrict__ path, unsigned short* __restrict__ sigsA) {
  __shared__ float wpath[WLEN * C_AUG];          // 272
  __shared__ float dxs[C_AUG];                   // 17
  __shared__ float e2[C_AUG * C_AUG];            // 289
  __shared__ float c1[C_AUG];                    // 17
  __shared__ float c2[C_AUG * C_AUG];            // 289
  __shared__ float c3[C_AUG * C_AUG * C_AUG];    // 4913

  int tid = threadIdx.x;
  int n = blockIdx.x;                 // 0 .. B*NGROUP*NWIN-1
  int b = n / (NGROUP * NWIN);
  int rem = n % (NGROUP * NWIN);
  int g = rem / NWIN;
  int w = rem % NWIN;
  int bg = b * NGROUP + g;
  int s0 = w * WSTEP;

  if (tid < WLEN * C_AUG)
    wpath[tid] = path[((size_t)bg * S_SZ + s0) * C_AUG + tid];
  __syncthreads();

  // --- init from dx0 ---
  if (tid < C_AUG) dxs[tid] = wpath[C_AUG + tid] - wpath[tid];
  __syncthreads();
  if (tid < C_AUG) c1[tid] = dxs[tid];
  if (tid < C_AUG * C_AUG) {
    int i = tid / C_AUG, j = tid % C_AUG;
    c2[tid] = dxs[i] * dxs[j] * 0.5f;
  }
  __syncthreads();
  for (int idx = tid; idx < C_AUG * C_AUG * C_AUG; idx += 256) {
    int ij = idx / C_AUG, k = idx % C_AUG;
    c3[idx] = c2[ij] * dxs[k] * (1.0f / 3.0f);
  }
  __syncthreads();

  // --- Chen steps for dx1..dx14 ---
  for (int t = 1; t < WLEN - 1; ++t) {
    if (tid < C_AUG)
      dxs[tid] = wpath[(t + 1) * C_AUG + tid] - wpath[t * C_AUG + tid];
    __syncthreads();
    if (tid < C_AUG * C_AUG) {
      int i = tid / C_AUG, j = tid % C_AUG;
      e2[tid] = dxs[i] * dxs[j] * 0.5f;
    }
    __syncthreads();
    // new3 = c3 + e3 + c1 (x) e2 + c2 (x) e1   (reads OLD c1,c2)
    for (int idx = tid; idx < C_AUG * C_AUG * C_AUG; idx += 256) {
      int i  = idx / (C_AUG * C_AUG);
      int jk = idx % (C_AUG * C_AUG);
      int ij = idx / C_AUG;
      int k  = idx % C_AUG;
      c3[idx] += (e2[ij] * (1.0f / 3.0f) + c2[ij]) * dxs[k] + c1[i] * e2[jk];
    }
    __syncthreads();
    // new2 = c2 + e2 + c1 (x) e1   (reads OLD c1)
    if (tid < C_AUG * C_AUG) {
      int i = tid / C_AUG, j = tid % C_AUG;
      c2[tid] += e2[tid] + c1[i] * dxs[j];
    }
    __syncthreads();
    if (tid < C_AUG) c1[tid] += dxs[tid];
    __syncthreads();
  }

  // --- emit concat(c1,c2,c3) as bf16 ---
  size_t base = (size_t)(w * B_SZ + b) * KPAD + (size_t)g * SIGC;
  for (int idx = tid; idx < SIGC; idx += 256) {
    float v = (idx < C_AUG) ? c1[idx]
            : (idx < C_AUG + C_AUG * C_AUG) ? c2[idx - C_AUG]
            : c3[idx - C_AUG - C_AUG * C_AUG];
    sigsA[base + idx] = f2bf(v);
  }
}

// ---------------------------------------------------------------------------
// 5) gi = sigs @ W_ih^T + b_ih.  M=2016, N=1536, K=10464 (bf16, f32 acc).
//    8 waves/block, each wave owns a 16Mx64N strip (4 accumulators).
//    Register double-buffered software pipeline: tile loads for K-step k+1
//    are issued before the WMMAs of step k, so the compiler emits partial
//    s_wait_loadcnt instead of draining to zero before every WMMA.
// ---------------------------------------------------------------------------
__global__ __launch_bounds__(256) void gemm_gi_kernel(
    const unsigned short* __restrict__ A,   // [MSIG][KPAD] bf16
    const unsigned short* __restrict__ Bw,  // [NGATE][KPAD] bf16
    const float* __restrict__ b_ih,
    float* __restrict__ gi) {               // [MSIG][NGATE] f32
  int wid  = blockIdx.x * 8 + (threadIdx.x >> 5);
  int lane = threadIdx.x & 31;
  int mtile  = wid / 24;            // 0..125
  int ngroup = wid % 24;            // 0..23  (64 cols each)
  int mbase = mtile * 16;
  int nbase = ngroup * 64;
  int l15 = lane & 15;
  int hi  = lane >> 4;

  const unsigned short* Arow = A + (size_t)(mbase + l15) * KPAD + (hi << 3);
  const unsigned short* B0   = Bw + (size_t)(nbase + l15) * KPAD + (hi << 3);

  v8f acc0 = {}, acc1 = {}, acc2 = {}, acc3 = {};

  V16 aC, b0C, b1C, b2C, b3C;     // stage "current"
  V16 aN, b0N, b1N, b2N, b3N;     // stage "next"

  auto loadStage = [&](V16& a, V16& b0, V16& b1, V16& b2, V16& b3, int kb) {
    a.u4[0]  = *(const uint4*)(Arow + kb);
    a.u4[1]  = *(const uint4*)(Arow + kb + 16);
    b0.u4[0] = *(const uint4*)(B0 + kb);
    b0.u4[1] = *(const uint4*)(B0 + kb + 16);
    b1.u4[0] = *(const uint4*)(B0 + (size_t)16 * KPAD + kb);
    b1.u4[1] = *(const uint4*)(B0 + (size_t)16 * KPAD + kb + 16);
    b2.u4[0] = *(const uint4*)(B0 + (size_t)32 * KPAD + kb);
    b2.u4[1] = *(const uint4*)(B0 + (size_t)32 * KPAD + kb + 16);
    b3.u4[0] = *(const uint4*)(B0 + (size_t)48 * KPAD + kb);
    b3.u4[1] = *(const uint4*)(B0 + (size_t)48 * KPAD + kb + 16);
  };
  auto mm = [&](V16& a, V16& b0, V16& b1, V16& b2, V16& b3) {
    acc0 = __builtin_amdgcn_wmma_f32_16x16x32_bf16(false, a.bf, false, b0.bf,
                                                   (short)0, acc0, false, false);
    acc1 = __builtin_amdgcn_wmma_f32_16x16x32_bf16(false, a.bf, false, b1.bf,
                                                   (short)0, acc1, false, false);
    acc2 = __builtin_amdgcn_wmma_f32_16x16x32_bf16(false, a.bf, false, b2.bf,
                                                   (short)0, acc2, false, false);
    acc3 = __builtin_amdgcn_wmma_f32_16x16x32_bf16(false, a.bf, false, b3.bf,
                                                   (short)0, acc3, false, false);
  };

  loadStage(aC, b0C, b1C, b2C, b3C, 0);
  // KPAD/32 == 327 (odd): pipelined double-stage loop covers pairs
  // (kb, kb+32); the final 32-chunk (KPAD-32) is left in the C stage.
#pragma unroll 1
  for (int kb = 0; kb + 64 <= KPAD; kb += 64) {
    loadStage(aN, b0N, b1N, b2N, b3N, kb + 32);     // issue next loads first
    __builtin_prefetch(Arow + kb + 512, 0, 1);
    mm(aC, b0C, b1C, b2C, b3C);                     // consume current
    if (kb + 64 < KPAD)
      loadStage(aC, b0C, b1C, b2C, b3C, kb + 64);   // refill current
    mm(aN, b0N, b1N, b2N, b3N);                     // consume next
  }
  mm(aC, b0C, b1C, b2C, b3C);                       // tail chunk (KPAD-32)

  int rbase = mbase + (hi << 3);
  int col0 = nbase + l15;
  float bb0 = b_ih[col0], bb1 = b_ih[col0 + 16];
  float bb2 = b_ih[col0 + 32], bb3 = b_ih[col0 + 48];
#pragma unroll
  for (int r = 0; r < 8; ++r) {
    size_t row = (size_t)(rbase + r) * NGATE;
    gi[row + col0]      = acc0[r] + bb0;
    gi[row + col0 + 16] = acc1[r] + bb1;
    gi[row + col0 + 32] = acc2[r] + bb2;
    gi[row + col0 + 48] = acc3[r] + bb3;
  }
}

// ---------------------------------------------------------------------------
// 6) GRU scan: one persistent 1024-thread block (32 waves). h (32x512 f32) in
//    LDS; per step gh = h @ W_hh^T via WMMA (each wave: 2 column tiles x 3
//    gate blocks -> gates fused in registers), then masked h update.
//    Ends with out = h @ lin_W^T + lin_b (waves 0..3).
// ---------------------------------------------------------------------------
__global__ __launch_bounds__(1024) void gru_kernel(
    const float* __restrict__ gi,            // [NWIN*B][1536]
    const unsigned short* __restrict__ Whh,  // [1536][512] bf16
    const float* __restrict__ b_hh,          // [1536]
    const int* __restrict__ lengths,         // [32]
    const unsigned short* __restrict__ linW, // [32][512] bf16
    const float* __restrict__ lin_b,         // [32]
    float* __restrict__ out) {               // [32][32]
  __shared__ float hsh[B_SZ * RNN];          // 64 KB
  int tid = threadIdx.x;
  for (int i = tid; i < B_SZ * RNN; i += 1024) hsh[i] = 0.0f;
  __syncthreads();

  int wave = tid >> 5, lane = tid & 31;
  int l15 = lane & 15;
  int hi  = lane >> 4;
  int mtile = wave >> 4;            // 0..1  (batch rows 0..15 / 16..31)
  int nb2   = (wave & 15) * 2;      // column-tile pair within a gate block
  int arow  = mtile * 16 + l15;
  v8f vzero = {};

  for (int w = 0; w < NWIN; ++w) {
    v8f acc[2][3];
#pragma unroll
    for (int jj = 0; jj < 2; ++jj)
#pragma unroll
      for (int gg = 0; gg < 3; ++gg) acc[jj][gg] = vzero;

#pragma unroll 1
    for (int kb = 0; kb < RNN; kb += 32) {
      V16 av;
      int koff = kb + (hi << 3);
#pragma unroll
      for (int e = 0; e < 8; ++e) av.us[e] = f2bf(hsh[arow * RNN + koff + e]);
#pragma unroll
      for (int e = 0; e < 8; ++e) av.us[8 + e] = f2bf(hsh[arow * RNN + koff + 16 + e]);
#pragma unroll
      for (int jj = 0; jj < 2; ++jj) {
#pragma unroll
        for (int gg = 0; gg < 3; ++gg) {
          int ncol = (gg * 32 + nb2 + jj) * 16 + l15;
          const unsigned short* Brow = Whh + (size_t)ncol * RNN + (hi << 3) + kb;
          V16 bv;
          bv.u4[0] = *(const uint4*)(Brow);
          bv.u4[1] = *(const uint4*)(Brow + 16);
          acc[jj][gg] = __builtin_amdgcn_wmma_f32_16x16x32_bf16(
              false, av.bf, false, bv.bf, (short)0, acc[jj][gg], false, false);
        }
      }
    }
    __syncthreads();   // all reads of hsh done before writes

#pragma unroll
    for (int jj = 0; jj < 2; ++jj) {
      int n = (nb2 + jj) * 16 + l15;  // hidden index 0..511
      float bhr = b_hh[n], bhz = b_hh[RNN + n], bhn = b_hh[2 * RNN + n];
#pragma unroll
      for (int r = 0; r < 8; ++r) {
        int m = mtile * 16 + (hi << 3) + r;       // batch index
        size_t grow = ((size_t)w * B_SZ + m) * NGATE;
        float ir = gi[grow + n];
        float iz = gi[grow + RNN + n];
        float inn = gi[grow + 2 * RNN + n];
        float hr = acc[jj][0][r] + bhr;
        float hz = acc[jj][1][r] + bhz;
        float hn = acc[jj][2][r] + bhn;
        float rg = 1.0f / (1.0f + __expf(-(ir + hr)));
        float zg = 1.0f / (1.0f + __expf(-(iz + hz)));
        float ng = tanhf(inn + rg * hn);
        float hold = hsh[m * RNN + n];
        float y = (1.0f - zg) * ng + zg * hold;
        bool msk = (w * WSTEP + WLEN) > lengths[m];
        hsh[m * RNN + n] = msk ? hold : y;
      }
    }
    __syncthreads();
  }

  // final linear (M=32, N=32, K=512): waves 0..3, one 16x16 tile each
  if (wave < 4) {
    int mt = wave >> 1, nt = wave & 1;
    int fr = mt * 16 + l15;
    v8f acc = vzero;
#pragma unroll 1
    for (int kb = 0; kb < RNN; kb += 32) {
      V16 av;
      int koff = kb + (hi << 3);
#pragma unroll
      for (int e = 0; e < 8; ++e) av.us[e] = f2bf(hsh[fr * RNN + koff + e]);
#pragma unroll
      for (int e = 0; e < 8; ++e) av.us[8 + e] = f2bf(hsh[fr * RNN + koff + 16 + e]);
      const unsigned short* Brow = linW + (size_t)(nt * 16 + l15) * RNN + (hi << 3) + kb;
      V16 bv;
      bv.u4[0] = *(const uint4*)(Brow);
      bv.u4[1] = *(const uint4*)(Brow + 16);
      acc = __builtin_amdgcn_wmma_f32_16x16x32_bf16(false, av.bf, false, bv.bf,
                                                    (short)0, acc, false, false);
    }
    int col = nt * 16 + l15;
#pragma unroll
    for (int r = 0; r < 8; ++r) {
      int row = mt * 16 + (hi << 3) + r;
      out[row * OUT_SZ + col] = acc[r] + lin_b[col];
    }
  }
}

// ---------------------------------------------------------------------------
extern "C" void kernel_launch(void* const* d_in, const int* in_sizes, int n_in,
                              void* d_out, int out_size, void* d_ws, size_t ws_size,
                              hipStream_t stream) {
  (void)in_sizes; (void)n_in; (void)out_size; (void)ws_size;
  const float* x      = (const float*)d_in[0];
  const int*   lens   = (const int*)d_in[1];
  const float* aug_W  = (const float*)d_in[2];
  const float* aug_b  = (const float*)d_in[3];
  const float* W_ih   = (const float*)d_in[4];
  const float* W_hh   = (const float*)d_in[5];
  const float* b_ih   = (const float*)d_in[6];
  const float* b_hh   = (const float*)d_in[7];
  const float* lin_W  = (const float*)d_in[8];
  const float* lin_b  = (const float*)d_in[9];
  float* out = (float*)d_out;

  char* ws = (char*)d_ws;
  size_t off = 0;
  auto take = [&](size_t bytes) {
    void* p = ws + off;
    off += (bytes + 255) & ~(size_t)255;
    return p;
  };
  float*          path  = (float*)take((size_t)NBG * S_SZ * C_AUG * sizeof(float));
  unsigned short* sigsA = (unsigned short*)take((size_t)MSIG * KPAD * 2);
  unsigned short* WihB  = (unsigned short*)take((size_t)NGATE * KPAD * 2);
  float*          gi    = (float*)take((size_t)MSIG * NGATE * sizeof(float));
  unsigned short* WhhB  = (unsigned short*)take((size_t)NGATE * RNN * 2);
  unsigned short* linWB = (unsigned short*)take((size_t)OUT_SZ * RNN * 2);

  path_kernel<<<(NBG * S_SZ + 255) / 256, 256, 0, stream>>>(x, lens, aug_W, aug_b, path);
  cvt_pad_kernel<<<2048, 256, 0, stream>>>(W_ih, WihB, NGATE, KSIG, KPAD);
  cvt_pad_kernel<<<1024, 256, 0, stream>>>(W_hh, WhhB, NGATE, RNN, RNN);
  cvt_pad_kernel<<<64, 256, 0, stream>>>(lin_W, linWB, OUT_SZ, RNN, RNN);
  sig_pad_kernel<<<(MSIG * (KPAD - KSIG) + 255) / 256, 256, 0, stream>>>(sigsA);
  sig_kernel<<<B_SZ * NGROUP * NWIN, 256, 0, stream>>>(path, sigsA);
  gemm_gi_kernel<<<(126 * 24) / 8, 256, 0, stream>>>(sigsA, WihB, b_ih, gi);
  gru_kernel<<<1, 1024, 0, stream>>>(gi, WhhB, b_hh, lens, linWB, lin_b, out);
}
